// Seq2SeqTranslation_56573309223225
// MI455X (gfx1250) — matrix operand order repaired and checked by reading, not compiled, not run on hardware
//
#include <hip/hip_runtime.h>

// ---------------------------------------------------------------------------
// Seq2Seq translation forward (encoder 2xLSTM -> context -> decoder 2xLSTM ->
// tanh proj -> vocab logits), CDNA5 / gfx1250.
// GEMMs: TDM (tensor_load_to_lds) double-buffered LDS staging + WMMA f16.
// Recurrent scan: persistent kernel, software grid barrier, direct-load WMMA.
// ---------------------------------------------------------------------------

typedef __attribute__((ext_vector_type(16))) _Float16 v16h;
typedef __attribute__((ext_vector_type(8)))  _Float16 v8h;
typedef __attribute__((ext_vector_type(8)))  float    v8f;
typedef __attribute__((ext_vector_type(4)))  unsigned int u32x4;
typedef __attribute__((ext_vector_type(4)))  int      i32x4;
typedef __attribute__((ext_vector_type(8)))  int      i32x8;

#define Bb   32
#define Ss   64
#define Tt   64
#define Ee   256
#define Hh   512
#define G4   2048      // 4*H
#define Vt   32000
#define NWG_REC 16     // workgroups in persistent recurrent kernel

#define KC       64    // K-chunk per TDM tile
#define LDS_LDA  72    // 64 f16 + 8 f16 TDM row padding (bank stagger, 16B aligned)

__device__ __forceinline__ float sigf(float x) { return 1.0f / (1.0f + __expf(-x)); }

// ---- WMMA fragment loaders (global, row-major) -----------------------------
// A 16x32 f16: lanes 0-15 row M=lane (K 0-15), lanes 16-31 (K 16-31)
__device__ __forceinline__ v16h frag_a(const _Float16* __restrict__ A, int lda,
                                       int row0, int kb, int lane) {
  const int r = row0 + (lane & 15);
  const int s = lane >> 4;
  const _Float16* p = A + (size_t)r * lda + kb + s * 16;
  v8h lo = *(const v8h*)p;
  v8h hi = *(const v8h*)(p + 8);
  return __builtin_shufflevector(lo, hi, 0,1,2,3,4,5,6,7,8,9,10,11,12,13,14,15);
}
// B 32x16 f16 from W[N,K] row-major: lane holds col n -> contiguous 16-K run
__device__ __forceinline__ v16h frag_b(const _Float16* __restrict__ W, int ldw,
                                       int n0, int kb, int lane) {
  const int n = n0 + (lane & 15);
  const int s = lane >> 4;
  return *(const v16h*)(W + (size_t)n * ldw + kb + s * 16);
}

// ---- WMMA fragment loaders (LDS tiles with stride LDS_LDA) -----------------
__device__ __forceinline__ v16h frag_a_lds(const _Float16* base, int row0, int kl, int lane) {
  const int r = row0 + (lane & 15);
  const int s = lane >> 4;
  const _Float16* p = base + r * LDS_LDA + kl + s * 16;
  v8h lo = *(const v8h*)p;          // ds_load_b128
  v8h hi = *(const v8h*)(p + 8);
  return __builtin_shufflevector(lo, hi, 0,1,2,3,4,5,6,7,8,9,10,11,12,13,14,15);
}
__device__ __forceinline__ v16h frag_b_lds(const _Float16* base, int n0, int kl, int lane) {
  const int n = n0 + (lane & 15);
  const int s = lane >> 4;
  const _Float16* p = base + n * LDS_LDA + kl + s * 16;
  v8h lo = *(const v8h*)p;
  v8h hi = *(const v8h*)(p + 8);
  return __builtin_shufflevector(lo, hi, 0,1,2,3,4,5,6,7,8,9,10,11,12,13,14,15);
}

// ---- Tensor Data Mover: 2D tile (tile_d0 x tile_d1) f16 -> LDS -------------
// D# per CDNA5 ISA. Padding: after every 32 DWORDs
// (one 64-elem row) insert 4 DWORDs -> LDS row stride = 72 f16 = 144B.
__device__ __forceinline__ void tdm_load_2d(unsigned lds_off, const _Float16* gptr,
                                            unsigned stride_elems) {
  const unsigned long long ga = (unsigned long long)(size_t)gptr;
  u32x4 g0;
  g0[0] = 1u;                                        // count=1 (valid user desc)
  g0[1] = lds_off;                                   // lds_addr (bytes)
  g0[2] = (unsigned)ga;                              // global_addr[31:0]
  g0[3] = (unsigned)((ga >> 32) & 0x01FFFFFFu) | (2u << 30);  // addr[56:32] | type=2
  i32x8 g1;
  g1[0] = (int)((1u << 16)        // data_size = 2 bytes
              | (1u << 20)        // pad_enable
              | (4u << 22)        // pad_interval: 32 DWORDs
              | (3u << 25));      // pad_amount: 4 DWORDs
  g1[1] = (int)((0x100000u & 0xFFFFu) << 16);        // tensor_dim0 lo16 (large)
  g1[2] = (int)((0x100000u >> 16) | ((0x100000u & 0xFFFFu) << 16)); // td0 hi / td1 lo
  g1[3] = (int)((0x100000u >> 16) | ((unsigned)KC << 16));          // td1 hi / tile_dim0
  g1[4] = (int)128u;                                 // tile_dim1 = 128 rows, tile_dim2 = 0
  g1[5] = (int)stride_elems;                         // tensor_dim0_stride lo32
  g1[6] = 0;                                         // stride hi16 / dim1_stride lo
  g1[7] = 0;
  i32x4 z4 = {0, 0, 0, 0};
#if __clang_major__ >= 23
  i32x8 z8 = {0, 0, 0, 0, 0, 0, 0, 0};
  __builtin_amdgcn_tensor_load_to_lds(g0, g1, z4, z4, z8, 0);
#else
  __builtin_amdgcn_tensor_load_to_lds(g0, g1, z4, z4, 0);
#endif
}

// ---- GEMM: C[M,N] = A[M,K](f16) @ W[N,K](f16)^T + bias ---------------------
// Block tile 128x128; TDM stages A/B 128xKC tiles into double-buffered LDS.
// 8 waves (2x4), wave tile 64x32 (4 M-tiles x 2 N-tiles).
__global__ __launch_bounds__(256) void gemm_f16_wmma(
    const _Float16* __restrict__ A, const _Float16* __restrict__ W,
    const float* __restrict__ bias, float* __restrict__ outF,
    _Float16* __restrict__ outH, int M, int N, int K, int tanh_f16)
{
  __shared__ __align__(16) _Float16 smem[2][2][128 * LDS_LDA];  // [buf][A/B] 72KB

  const int lane = threadIdx.x & 31;
  const int wave = threadIdx.x >> 5;
  const int nb = N >> 7;
  const int bm = blockIdx.x / nb;
  const int bn = blockIdx.x - bm * nb;
  const int rowLoc = (wave >> 2) * 64;           // local row of wave tile
  const int colLoc = (wave & 3) * 32;            // local col of wave tile
  const int rowBase = bm * 128 + rowLoc;
  const int colBase = bn * 128 + colLoc;
  const _Float16* Ablk = A + (size_t)(bm * 128) * K;
  const _Float16* Wblk = W + (size_t)(bn * 128) * K;
  const int nk = K / KC;

  v8f acc[4][2];
#pragma unroll
  for (int i = 0; i < 4; ++i)
#pragma unroll
    for (int j = 0; j < 2; ++j) acc[i][j] = (v8f)(0.0f);

  if (threadIdx.x == 0) {                         // prologue: chunk 0
    tdm_load_2d((unsigned)(size_t)&smem[0][0][0], Ablk, (unsigned)K);
    tdm_load_2d((unsigned)(size_t)&smem[0][1][0], Wblk, (unsigned)K);
  }

  for (int kc = 0; kc < nk; ++kc) {
    const int buf = kc & 1;
    if (threadIdx.x == 0) {
      if (kc + 1 < nk) {                          // issue next chunk, wait current
        tdm_load_2d((unsigned)(size_t)&smem[buf ^ 1][0][0],
                    Ablk + (kc + 1) * KC, (unsigned)K);
        tdm_load_2d((unsigned)(size_t)&smem[buf ^ 1][1][0],
                    Wblk + (kc + 1) * KC, (unsigned)K);
        __builtin_amdgcn_s_wait_tensorcnt(2);     // in-order TDM: chunk kc done
      } else {
        __builtin_amdgcn_s_wait_tensorcnt(0);
      }
    }
    __syncthreads();

    const _Float16* As = &smem[buf][0][0];
    const _Float16* Bs = &smem[buf][1][0];
#pragma unroll
    for (int ks = 0; ks < KC / 32; ++ks) {
      const int kl = ks * 32;
      v16h bf0 = frag_b_lds(Bs, colLoc,      kl, lane);
      v16h bf1 = frag_b_lds(Bs, colLoc + 16, kl, lane);
#pragma unroll
      for (int i = 0; i < 4; ++i) {
        v16h af = frag_a_lds(As, rowLoc + i * 16, kl, lane);
        acc[i][0] = __builtin_amdgcn_wmma_f32_16x16x32_f16(
            false, af, false, bf0, (short)0, acc[i][0], false, false);
        acc[i][1] = __builtin_amdgcn_wmma_f32_16x16x32_f16(
            false, af, false, bf1, (short)0, acc[i][1], false, false);
      }
    }
    __syncthreads();                              // buffer reuse fence
  }

  // C/D layout: VGPR r, lane L -> row = r + 8*(L>=16), col = L%16
  const int s = lane >> 4, ln = lane & 15;
#pragma unroll
  for (int j = 0; j < 2; ++j) {
    const int col = colBase + j * 16 + ln;
    const float bv = bias[col];
#pragma unroll
    for (int i = 0; i < 4; ++i) {
#pragma unroll
      for (int r = 0; r < 8; ++r) {
        const int row = rowBase + i * 16 + s * 8 + r;
        const float v = acc[i][j][r] + bv;
        const size_t idx = (size_t)row * N + col;
        if (tanh_f16) outH[idx] = (_Float16)tanhf(v);
        else          outF[idx] = v;
      }
    }
  }
}

// ---- software grid barrier for the persistent recurrent kernel -------------
__device__ __forceinline__ void grid_sync(unsigned* bar, unsigned target) {
  __syncthreads();
  if (threadIdx.x == 0) {
    __hip_atomic_fetch_add(bar, 1u, __ATOMIC_ACQ_REL, __HIP_MEMORY_SCOPE_AGENT);
    while (__hip_atomic_load(bar, __ATOMIC_ACQUIRE, __HIP_MEMORY_SCOPE_AGENT) < target)
      __builtin_amdgcn_s_sleep(2);
  }
  __syncthreads();
}

// ---- persistent LSTM scan: 64 steps, gates = xW[:,t,:] + h @ Whh^T ---------
__global__ __launch_bounds__(256) void lstm_rec(
    const float* __restrict__ xW,        // [B,T,4H] f32  (row b*T+t)
    const _Float16* __restrict__ Whh,    // [4H,H] f16
    const float* __restrict__ h0,        // [B,H] init h (f32)
    const float* __restrict__ c0,        // [B,H] init c (f32)
    _Float16* __restrict__ hs,           // [B,T,H] f16 outputs
    float* __restrict__ hT,              // [B,H] final h (f32)
    float* __restrict__ cC,              // [B,H] working c == final c
    float* __restrict__ gates,           // [B,4H] f32 scratch
    _Float16* __restrict__ hcur,         // [B,H] f16 working h
    unsigned* __restrict__ bar)
{
  for (int e = blockIdx.x * 256 + threadIdx.x; e < Bb * Hh; e += NWG_REC * 256) {
    hcur[e] = (_Float16)h0[e];
    cC[e]   = c0[e];
  }
  unsigned target = NWG_REC;
  grid_sync(bar, target);

  const int lane  = threadIdx.x & 31;
  const int wave  = threadIdx.x >> 5;
  const int nTile = blockIdx.x * 128 + wave * 16;  // this wave's 16 gate cols
  const int s = lane >> 4, ln = lane & 15;
  const int jW = Hh / NWG_REC;                     // 32 h-cols per WG
  const int jBase = blockIdx.x * jW;

  for (int t = 0; t < Tt; ++t) {
    v8f acc0 = (v8f)(0.0f), acc1 = (v8f)(0.0f);
    for (int kb = 0; kb < Hh; kb += 32) {
      v16h bf = frag_b(Whh, Hh, nTile, kb, lane);
      v16h a0 = frag_a(hcur, Hh, 0,  kb, lane);
      v16h a1 = frag_a(hcur, Hh, 16, kb, lane);
      acc0 = __builtin_amdgcn_wmma_f32_16x16x32_f16(
          false, a0, false, bf, (short)0, acc0, false, false);
      acc1 = __builtin_amdgcn_wmma_f32_16x16x32_f16(
          false, a1, false, bf, (short)0, acc1, false, false);
    }
    const int col = nTile + ln;
#pragma unroll
    for (int r = 0; r < 8; ++r) {
      const int b0i = s * 8 + r;
      gates[(size_t)b0i * G4 + col] = acc0[r] + xW[((size_t)b0i * Tt + t) * G4 + col];
      const int b1i = 16 + s * 8 + r;
      gates[(size_t)b1i * G4 + col] = acc1[r] + xW[((size_t)b1i * Tt + t) * G4 + col];
    }
    target += NWG_REC;
    grid_sync(bar, target);

    for (int e = threadIdx.x; e < Bb * jW; e += 256) {
      const int b = e / jW;
      const int j = jBase + (e - b * jW);
      const float ig = sigf (gates[(size_t)b * G4 +          j]);
      const float fg = sigf (gates[(size_t)b * G4 + Hh     + j]);
      const float gg = tanhf(gates[(size_t)b * G4 + 2 * Hh + j]);
      const float og = sigf (gates[(size_t)b * G4 + 3 * Hh + j]);
      const float c  = fg * cC[b * Hh + j] + ig * gg;
      const float h  = og * tanhf(c);
      cC[b * Hh + j]   = c;
      hcur[b * Hh + j] = (_Float16)h;
      hs[((size_t)b * Tt + t) * Hh + j] = (_Float16)h;
      if (t == Tt - 1) hT[b * Hh + j] = h;
    }
    target += NWG_REC;
    grid_sync(bar, target);
  }
}

// ---- small helper kernels --------------------------------------------------
__global__ void cvt_f16(const float* __restrict__ x, _Float16* __restrict__ y, int n) {
  int i = blockIdx.x * 256 + threadIdx.x;
  if (i < n) y[i] = (_Float16)x[i];
}

__global__ void embed_gather(const int* __restrict__ ids, const float* __restrict__ emb,
                             _Float16* __restrict__ out, int nrows, int width) {
  int i = blockIdx.x * 256 + threadIdx.x;
  if (i < nrows * width) {
    int r = i / width, c = i - r * width;
    out[i] = (_Float16)emb[(size_t)ids[r] * width + c];
  }
}

__global__ void ctx_mean(const _Float16* __restrict__ e1, float* __restrict__ ctx) {
  int i = blockIdx.x * 256 + threadIdx.x;    // B*H elements
  if (i < Bb * Hh) {
    int b = i / Hh, j = i - b * Hh;
    float acc = 0.0f;
    for (int st = 0; st < Ss; ++st)
      acc += (float)e1[((size_t)b * Ss + st) * Hh + j];
    ctx[i] = acc * (1.0f / Ss);
  }
}

__global__ void build_dec_in(const int* __restrict__ tgt, const float* __restrict__ emb,
                             const float* __restrict__ ctx, _Float16* __restrict__ out) {
  int i = blockIdx.x * 256 + threadIdx.x;    // B*T*(E+H)
  const int W = Ee + Hh;
  if (i < Bb * Tt * W) {
    int row = i / W, c = i - row * W;
    float v;
    if (c < Ee) v = emb[(size_t)tgt[row] * Ee + c];
    else        v = ctx[(row / Tt) * Hh + (c - Ee)];
    out[i] = (_Float16)v;
  }
}

// ---------------------------------------------------------------------------
extern "C" void kernel_launch(void* const* d_in, const int* in_sizes, int n_in,
                              void* d_out, int out_size, void* d_ws, size_t ws_size,
                              hipStream_t stream)
{
  const int*   src     = (const int*)  d_in[0];
  const int*   tgt     = (const int*)  d_in[1];
  const float* enc_emb = (const float*)d_in[2];
  const float* dec_emb = (const float*)d_in[3];
  const float* eWih0 = (const float*)d_in[4];
  const float* eWhh0 = (const float*)d_in[5];
  const float* eb0   = (const float*)d_in[6];
  const float* eWih1 = (const float*)d_in[7];
  const float* eWhh1 = (const float*)d_in[8];
  const float* eb1   = (const float*)d_in[9];
  const float* dWih0 = (const float*)d_in[10];
  const float* dWhh0 = (const float*)d_in[11];
  const float* db0   = (const float*)d_in[12];
  const float* dWih1 = (const float*)d_in[13];
  const float* dWhh1 = (const float*)d_in[14];
  const float* db1   = (const float*)d_in[15];
  const float* pW1   = (const float*)d_in[16];
  const float* pb1   = (const float*)d_in[17];
  const float* pW2   = (const float*)d_in[18];
  const float* pb2   = (const float*)d_in[19];
  float* logits = (float*)d_out;

  // workspace bump allocator (256B aligned)
  char* p = (char*)d_ws;
  auto alloc = [&](size_t bytes) -> char* {
    char* r = p; p += (bytes + 255) & ~(size_t)255; return r;
  };
  unsigned* bar  = (unsigned*)alloc(256);               // 4 barrier counters (64B apart)
  float*    zeros = (float*)alloc(sizeof(float) * Bb * Hh);
  const size_t zlen = (size_t)(p - (char*)d_ws);        // zero bar + zeros every call

  // f16 weight copies
  _Float16* hWih0e = (_Float16*)alloc(2 * (size_t)G4 * Ee);
  _Float16* hWhh0e = (_Float16*)alloc(2 * (size_t)G4 * Hh);
  _Float16* hWih1e = (_Float16*)alloc(2 * (size_t)G4 * Hh);
  _Float16* hWhh1e = (_Float16*)alloc(2 * (size_t)G4 * Hh);
  _Float16* hWih0d = (_Float16*)alloc(2 * (size_t)G4 * (Ee + Hh));
  _Float16* hWhh0d = (_Float16*)alloc(2 * (size_t)G4 * Hh);
  _Float16* hWih1d = (_Float16*)alloc(2 * (size_t)G4 * Hh);
  _Float16* hWhh1d = (_Float16*)alloc(2 * (size_t)G4 * Hh);
  _Float16* hW1    = (_Float16*)alloc(2 * (size_t)Hh * Hh);
  _Float16* hW2    = (_Float16*)alloc(2 * (size_t)Vt * Hh);
  // activations
  _Float16* xe   = (_Float16*)alloc(2 * (size_t)Bb * Ss * Ee);
  _Float16* rnn  = (_Float16*)alloc(2 * (size_t)Bb * Tt * (Ee + Hh));
  _Float16* e0   = (_Float16*)alloc(2 * (size_t)Bb * Ss * Hh);
  _Float16* e1   = (_Float16*)alloc(2 * (size_t)Bb * Ss * Hh);
  _Float16* d0   = (_Float16*)alloc(2 * (size_t)Bb * Tt * Hh);
  _Float16* d1   = (_Float16*)alloc(2 * (size_t)Bb * Tt * Hh);
  _Float16* hid  = (_Float16*)alloc(2 * (size_t)Bb * Tt * Hh);
  _Float16* hcur = (_Float16*)alloc(2 * (size_t)Bb * Hh);
  float* xW    = (float*)alloc(4 * (size_t)Bb * Tt * G4);
  float* gates = (float*)alloc(4 * (size_t)Bb * G4);
  float* cT0   = (float*)alloc(4 * (size_t)Bb * Hh);
  float* cT1   = (float*)alloc(4 * (size_t)Bb * Hh);
  float* cTd0  = (float*)alloc(4 * (size_t)Bb * Hh);
  float* cTd1  = (float*)alloc(4 * (size_t)Bb * Hh);
  float* hT0   = (float*)alloc(4 * (size_t)Bb * Hh);
  float* hT1   = (float*)alloc(4 * (size_t)Bb * Hh);
  float* hTd   = (float*)alloc(4 * (size_t)Bb * Hh);
  float* ctx   = (float*)alloc(4 * (size_t)Bb * Hh);

  hipMemsetAsync(d_ws, 0, zlen, stream);  // barriers + zero initial states

  auto cvt = [&](const float* s_, _Float16* d_, int n) {
    cvt_f16<<<(n + 255) / 256, 256, 0, stream>>>(s_, d_, n);
  };
  cvt(eWih0, hWih0e, G4 * Ee);
  cvt(eWhh0, hWhh0e, G4 * Hh);
  cvt(eWih1, hWih1e, G4 * Hh);
  cvt(eWhh1, hWhh1e, G4 * Hh);
  cvt(dWih0, hWih0d, G4 * (Ee + Hh));
  cvt(dWhh0, hWhh0d, G4 * Hh);
  cvt(dWih1, hWih1d, G4 * Hh);
  cvt(dWhh1, hWhh1d, G4 * Hh);
  cvt(pW1, hW1, Hh * Hh);
  cvt(pW2, hW2, Vt * Hh);

  auto gemm = [&](const _Float16* A, const _Float16* W, const float* bias,
                  float* oF, _Float16* oH, int M, int N, int K, int mode) {
    const int grid = (M / 128) * (N / 128);
    gemm_f16_wmma<<<grid, 256, 0, stream>>>(A, W, bias, oF, oH, M, N, K, mode);
  };

  // ---- encoder ----
  {
    const int n = Bb * Ss * Ee;
    embed_gather<<<(n + 255) / 256, 256, 0, stream>>>(src, enc_emb, xe, Bb * Ss, Ee);
  }
  gemm(xe, hWih0e, eb0, xW, nullptr, Bb * Ss, G4, Ee, 0);
  lstm_rec<<<NWG_REC, 256, 0, stream>>>(xW, hWhh0e, zeros, zeros, e0, hT0, cT0,
                                        gates, hcur, bar + 0);
  gemm(e0, hWih1e, eb1, xW, nullptr, Bb * Ss, G4, Hh, 0);
  lstm_rec<<<NWG_REC, 256, 0, stream>>>(xW, hWhh1e, zeros, zeros, e1, hT1, cT1,
                                        gates, hcur, bar + 16);

  // ---- context + decoder input ----
  {
    const int n = Bb * Hh;
    ctx_mean<<<(n + 255) / 256, 256, 0, stream>>>(e1, ctx);
    const int m = Bb * Tt * (Ee + Hh);
    build_dec_in<<<(m + 255) / 256, 256, 0, stream>>>(tgt, dec_emb, ctx, rnn);
  }

  // ---- decoder ----
  gemm(rnn, hWih0d, db0, xW, nullptr, Bb * Tt, G4, Ee + Hh, 0);
  lstm_rec<<<NWG_REC, 256, 0, stream>>>(xW, hWhh0d, hT0, cT0, d0, hTd, cTd0,
                                        gates, hcur, bar + 32);
  gemm(d0, hWih1d, db1, xW, nullptr, Bb * Tt, G4, Hh, 0);
  lstm_rec<<<NWG_REC, 256, 0, stream>>>(xW, hWhh1d, hT1, cT1, d1, hTd, cTd1,
                                        gates, hcur, bar + 48);

  // ---- output projection ----
  gemm(d1, hW1, pb1, nullptr, hid, Bb * Tt, Hh, Hh, 1);        // tanh -> f16
  gemm(hid, hW2, pb2, logits, nullptr, Bb * Tt, Vt, Hh, 0);    // logits f32
}
